// BaseLayerWithLoRAMultiStream_72980084293845
// MI455X (gfx1250) — matrix operand order repaired and checked by reading, not compiled, not run on hardware
//
#include <hip/hip_runtime.h>

// ---------------------------------------------------------------------------
// LoRA-fused linear layer for MI455X (gfx1250, wave32, WMMA bf16 16x16x32)
//   out[M,N] = x[M,K] @ W[N,K]^T + b[N] + (x @ A)[M,R] @ B[R,N]
//   M=8192 (B*S), K=4096, N=4096, R=16
//
// Kernel 2 is a software-pipelined, double-buffered WMMA GEMM:
//   iter k:  global_load(k+1) -> VGPRs | compute(buf k%2) | cvt+ds_store(buf (k+1)%2)
//   one s_barrier per iteration; LoRA handled as one zero-padded K=32 step.
// ---------------------------------------------------------------------------

typedef __attribute__((ext_vector_type(16))) __bf16 v16bf;
typedef __attribute__((ext_vector_type(2)))  __bf16 v2bf;
typedef __attribute__((ext_vector_type(8)))  float  v8f;

#define M_TOT 8192
#define K_TOT 4096
#define N_TOT 4096
#define RANK  16

#define BM 128
#define BN 128
#define BK 32
#define LDSS 40   // halves per LDS row: 32 data + 8 pad (80B = 16B-aligned, conflict-free)

// ---- fp32 -> packed bf16x2 ------------------------------------------------
// Plain fptrunc casts: ISel picks the native v_cvt(_pk)_bf16_f32 when the
// target has it, else expands to the RNE bit sequence. Zero compile risk.
__device__ __forceinline__ unsigned int pack2(float a, float b) {
  v2bf p;
  p.x = (__bf16)a;
  p.y = (__bf16)b;
  return __builtin_bit_cast(unsigned int, p);
}

union Frag { uint4 q[2]; v16bf v; };

// 16 contiguous fp32 held in VGPRs between fetch and LDS store (pipeline stage)
struct F16x { float4 v[4]; };

__device__ __forceinline__ F16x fetch16(const float* __restrict__ g) {
  const float4* g4 = (const float4*)g;
  F16x r;
  r.v[0] = g4[0]; r.v[1] = g4[1]; r.v[2] = g4[2]; r.v[3] = g4[3];
  return r;
}
__device__ __forceinline__ void cvt_store(const F16x& f, unsigned short* l) {
  uint4 q0, q1;
  q0.x = pack2(f.v[0].x, f.v[0].y); q0.y = pack2(f.v[0].z, f.v[0].w);
  q0.z = pack2(f.v[1].x, f.v[1].y); q0.w = pack2(f.v[1].z, f.v[1].w);
  q1.x = pack2(f.v[2].x, f.v[2].y); q1.y = pack2(f.v[2].z, f.v[2].w);
  q1.z = pack2(f.v[3].x, f.v[3].y); q1.w = pack2(f.v[3].z, f.v[3].w);
  *(uint4*)(l)     = q0;
  *(uint4*)(l + 8) = q1;
}
__device__ __forceinline__ void stage_zero(unsigned short* l) {
  uint4 z; z.x = 0u; z.y = 0u; z.z = 0u; z.w = 0u;
  *(uint4*)(l)     = z;
  *(uint4*)(l + 8) = z;
}

// ---------------------------------------------------------------------------
// Kernel 1: t[M,R] = x[M,K] @ A[K,R]   (tiny: ~1 GFLOP, one block per row)
// ---------------------------------------------------------------------------
__global__ __launch_bounds__(256) void lora_xa_kernel(
    const float* __restrict__ x, const float* __restrict__ A,
    float* __restrict__ t) {
  __shared__ float red[256 * RANK];
  const int m   = blockIdx.x;
  const int tid = threadIdx.x;

  float acc[RANK];
#pragma unroll
  for (int r = 0; r < RANK; ++r) acc[r] = 0.f;

  const float* xrow = x + (size_t)m * K_TOT;
  for (int k = tid; k < K_TOT; k += 256) {
    float xv = xrow[k];
    const float* arow = A + (size_t)k * RANK;
#pragma unroll
    for (int r = 0; r < RANK; ++r) acc[r] = fmaf(xv, arow[r], acc[r]);
  }
#pragma unroll
  for (int r = 0; r < RANK; ++r) red[tid * RANK + r] = acc[r];

  for (int s = 128; s > 0; s >>= 1) {
    __syncthreads();
    if (tid < s) {
#pragma unroll
      for (int r = 0; r < RANK; ++r)
        red[tid * RANK + r] += red[(tid + s) * RANK + r];
    }
  }
  __syncthreads();
  if (tid < RANK) t[(size_t)m * RANK + tid] = red[tid];
}

// ---------------------------------------------------------------------------
// Kernel 2: pipelined double-buffered WMMA GEMM, fused bias + rank-16 K-ext
// ---------------------------------------------------------------------------
__global__ __launch_bounds__(256) void lora_gemm_kernel(
    const float* __restrict__ x, const float* __restrict__ W,
    const float* __restrict__ bias, const float* __restrict__ t,
    const float* __restrict__ loraB, float* __restrict__ out) {
  __shared__ unsigned short lds_a[2][BM * LDSS];   // x tiles,  bf16, [m][k]
  __shared__ unsigned short lds_b[2][BN * LDSS];   // W tiles,  bf16, [n][k]

  const int tid = threadIdx.x;
  const int n0  = blockIdx.x * BN;
  const int m0  = blockIdx.y * BM;

  const int lane   = tid & 31;
  const int wave   = tid >> 5;     // 0..7
  const int wave_m = wave >> 2;    // 0..1 -> 64 rows
  const int wave_n = wave & 3;     // 0..3 -> 32 cols
  const int lrow   = lane & 15;
  const int hsel   = lane >> 4;    // K-half select per ISA fragment layout

  const int srow = tid >> 1;          // staging row 0..127
  const int sseg = (tid & 1) * 16;    // staging k-offset 0 / 16
  const int soff = srow * LDSS + sseg;

  v8f acc[4][2];
#pragma unroll
  for (int i = 0; i < 4; ++i)
#pragma unroll
    for (int j = 0; j < 2; ++j)
#pragma unroll
      for (int e = 0; e < 8; ++e) acc[i][j][e] = 0.f;

  const float* xg = x + (size_t)(m0 + srow) * K_TOT + sseg;
  const float* wg = W + (size_t)(n0 + srow) * K_TOT + sseg;

  // one 16x16x32 WMMA step over the staged tiles (8 WMMA / wave)
  auto compute = [&](int buf) {
    Frag a[4], bfr[2];
#pragma unroll
    for (int fm = 0; fm < 4; ++fm) {
      const unsigned short* p =
          &lds_a[buf][(wave_m * 64 + fm * 16 + lrow) * LDSS];
      a[fm].q[0] = *(const uint4*)(p + hsel * 8);        // K {0..7} / {8..15}
      a[fm].q[1] = *(const uint4*)(p + 16 + hsel * 8);   // K {16..23} / {24..31}
    }
#pragma unroll
    for (int fn = 0; fn < 2; ++fn) {
      const unsigned short* p =
          &lds_b[buf][(wave_n * 32 + fn * 16 + lrow) * LDSS];
      bfr[fn].q[0] = *(const uint4*)(p + hsel * 8);        // K {0..7} / {8..15}
      bfr[fn].q[1] = *(const uint4*)(p + 16 + hsel * 8);   // K {16..23} / {24..31}
    }
#pragma unroll
    for (int fm = 0; fm < 4; ++fm)
#pragma unroll
      for (int fn = 0; fn < 2; ++fn)
        acc[fm][fn] = __builtin_amdgcn_wmma_f32_16x16x32_bf16(
            false, a[fm].v, false, bfr[fn].v, (short)0, acc[fm][fn],
            false, false);
  };

  const int KT = K_TOT / BK;   // 128

  // --- pipeline prologue: stage tile 0 into buffer 0 ---
  F16x fx = fetch16(xg);
  F16x fw = fetch16(wg);
  cvt_store(fx, &lds_a[0][soff]);
  cvt_store(fw, &lds_b[0][soff]);
  __syncthreads();

  // --- main loop: one barrier per iteration ---
  for (int kt = 0; kt < KT; ++kt) {
    const int cur = kt & 1;
    const int nxt = cur ^ 1;
    const bool have_next = (kt + 1 < KT);

    if (have_next) {                       // fetch k+1 while computing k
      fx = fetch16(xg + (size_t)(kt + 1) * BK);
      fw = fetch16(wg + (size_t)(kt + 1) * BK);
    }
    if (kt + 2 < KT) {                     // warm L2 two tiles ahead
      __builtin_prefetch(xg + (size_t)(kt + 2) * BK, 0, 3);
      __builtin_prefetch(wg + (size_t)(kt + 2) * BK, 0, 3);
    }

    compute(cur);

    if (have_next) {
      cvt_store(fx, &lds_a[nxt][soff]);
      cvt_store(fw, &lds_b[nxt][soff]);
    } else {
      // --- LoRA K-extension staged into the free buffer: A := t, B := lora_B
      if (sseg == 0) {
        F16x ft = fetch16(t + (size_t)(m0 + srow) * RANK);
        cvt_store(ft, &lds_a[nxt][soff]);
        float v[16];
#pragma unroll
        for (int k = 0; k < 16; ++k)
          v[k] = loraB[(size_t)k * N_TOT + (n0 + srow)];  // B[k][n] (one-time)
        uint4 q0, q1;
        q0.x = pack2(v[0],  v[1]);  q0.y = pack2(v[2],  v[3]);
        q0.z = pack2(v[4],  v[5]);  q0.w = pack2(v[6],  v[7]);
        q1.x = pack2(v[8],  v[9]);  q1.y = pack2(v[10], v[11]);
        q1.z = pack2(v[12], v[13]); q1.w = pack2(v[14], v[15]);
        unsigned short* lb = &lds_b[nxt][soff];
        *(uint4*)(lb)     = q0;
        *(uint4*)(lb + 8) = q1;
      } else {
        stage_zero(&lds_a[nxt][soff]);     // zero-pad K = 16..31
        stage_zero(&lds_b[nxt][soff]);
      }
    }
    __syncthreads();
  }

  compute(KT & 1);   // the LoRA step

  // --- epilogue: bias add + non-temporal store (out is write-once, 134 MB;
  //     keep L2 for W/x reuse) ---
#pragma unroll
  for (int fm = 0; fm < 4; ++fm) {
    const int rbase = m0 + wave_m * 64 + fm * 16 + hsel * 8;
#pragma unroll
    for (int fn = 0; fn < 2; ++fn) {
      const int col = n0 + wave_n * 32 + fn * 16 + lrow;
      const float bv = bias[col];
      float* op = out + (size_t)rbase * N_TOT + col;
#pragma unroll
      for (int e = 0; e < 8; ++e)
        __builtin_nontemporal_store(acc[fm][fn][e] + bv, op + (size_t)e * N_TOT);
    }
  }
}

// ---------------------------------------------------------------------------
extern "C" void kernel_launch(void* const* d_in, const int* in_sizes, int n_in,
                              void* d_out, int out_size, void* d_ws, size_t ws_size,
                              hipStream_t stream) {
  const float* x      = (const float*)d_in[0];   // [4,2048,4096]
  const float* W      = (const float*)d_in[1];   // [4096,4096]
  const float* bias   = (const float*)d_in[2];   // [4096]
  const float* lora_A = (const float*)d_in[3];   // [4096,16]
  const float* lora_B = (const float*)d_in[4];   // [16,4096]
  float* out = (float*)d_out;                    // [4,2048,4096]
  float* t   = (float*)d_ws;                     // [8192,16] fp32 (512 KB)

  (void)in_sizes; (void)n_in; (void)out_size; (void)ws_size;

  lora_xa_kernel<<<M_TOT, 256, 0, stream>>>(x, lora_A, t);

  dim3 grid(N_TOT / BN, M_TOT / BM);   // (32, 64)
  lora_gemm_kernel<<<grid, 256, 0, stream>>>(x, W, bias, t, lora_B, out);
}
